// CFSS_26233660244418
// MI455X (gfx1250) — matrix operand order repaired and checked by reading, not compiled, not run on hardware
//
#include <hip/hip_runtime.h>
#include <hip/hip_bf16.h>
#include <math.h>

typedef __attribute__((ext_vector_type(16))) _Float16 v16h;
typedef __attribute__((ext_vector_type(8)))  float    v8f;

#define B_SZ 2
#define HH 64
#define WW 64
#define LL 4096
#define DI 192
#define NS 16
#define RR 6
#define KK 4
#define CMID 38   /* R + 2N */

__device__ __forceinline__ float sigmoidf_(float x){ return 1.f/(1.f+__expf(-x)); }
__device__ __forceinline__ float siluf_(float x){ return x*sigmoidf_(x); }

// ---------------------------------------------------------------------------
// Fully-static strided, batched WMMA GEMM:  C[m,n] = sum_k A[m,k]*B[k,n]
// One wave computes a 32x32 C tile: 2 A-fragments x 2 B-fragments ->
// 4 v_wmma_f32_16x16x32_f16 per 32-wide k-step, fp32 accumulate.
// All shapes/strides are template constants -> 32-bit constant-stride address
// math, unconditional (mergeable) loads except where a mask is statically
// required (N%32!=0 or K%32!=0).
// ---------------------------------------------------------------------------
template<int M, int N, int Kd,
         int ASR, int ASC, int BSR, int BSC, int CSR, int CSC,
         int ABAT, int AMOD, int BBAT, int BMOD, int CBAT, int CMOD>
__global__ __launch_bounds__(32) void wmma_gemm(const float* __restrict__ A,
    const float* __restrict__ B, float* __restrict__ C)
{
  constexpr bool KMASK = (Kd % 32) != 0;
  constexpr bool NMASK = (N  % 32) != 0;
  int bz = blockIdx.z;
  const float* Ab = A + (bz % AMOD) * ABAT;
  const float* Bb = B + (bz % BMOD) * BBAT;
  float*       Cb = C + (bz % CMOD) * CBAT;
  int tm = blockIdx.y * 32, tn = blockIdx.x * 32;
  int lane = threadIdx.x & 31;
  int half = lane >> 4;
  int lr   = lane & 15;
  int am0 = (tm + lr)      * ASR;   // A row offsets (rows always in range: M%32==0)
  int am1 = (tm + 16 + lr) * ASR;
  int bn0 = tn + lr;                // B col indices
  int bn1 = tn + 16 + lr;
  v8f acc00 = {}, acc01 = {}, acc10 = {}, acc11 = {};
  for (int k0 = 0; k0 < Kd; k0 += 32) {
    v16h a0, a1, b0, b1;
#pragma unroll
    for (int e = 0; e < 16; ++e) {
      int ka = k0 + e + 8*((e>>3) + half);   // A element k (ISA swizzle)
      int kb = k0 + e + 8*((e>>3) + half);   // B element k
      if constexpr (KMASK) {
        int kac = ka < Kd ? ka : 0;
        int kbc = kb < Kd ? kb : 0;
        float av0 = Ab[am0 + kac*ASC];
        float av1 = Ab[am1 + kac*ASC];
        a0[e] = (_Float16)(ka < Kd ? av0 : 0.f);
        a1[e] = (_Float16)(ka < Kd ? av1 : 0.f);
        float bv0 = Bb[kbc*BSR + bn0*BSC];
        float bv1 = Bb[kbc*BSR + bn1*BSC];
        b0[e] = (_Float16)(kb < Kd ? bv0 : 0.f);
        b1[e] = (_Float16)(kb < Kd ? bv1 : 0.f);
      } else {
        a0[e] = (_Float16)Ab[am0 + ka*ASC];
        a1[e] = (_Float16)Ab[am1 + ka*ASC];
        if constexpr (NMASK) {
          int n0c = bn0 < N ? bn0 : 0;
          int n1c = bn1 < N ? bn1 : 0;
          float bv0 = Bb[kb*BSR + n0c*BSC];
          float bv1 = Bb[kb*BSR + n1c*BSC];
          b0[e] = (_Float16)(bn0 < N ? bv0 : 0.f);
          b1[e] = (_Float16)(bn1 < N ? bv1 : 0.f);
        } else {
          b0[e] = (_Float16)Bb[kb*BSR + bn0*BSC];
          b1[e] = (_Float16)Bb[kb*BSR + bn1*BSC];
        }
      }
    }
    acc00 = __builtin_amdgcn_wmma_f32_16x16x32_f16(false, a0, false, b0, (short)0, acc00, false, false);
    acc01 = __builtin_amdgcn_wmma_f32_16x16x32_f16(false, a0, false, b1, (short)0, acc01, false, false);
    acc10 = __builtin_amdgcn_wmma_f32_16x16x32_f16(false, a1, false, b0, (short)0, acc10, false, false);
    acc11 = __builtin_amdgcn_wmma_f32_16x16x32_f16(false, a1, false, b1, (short)0, acc11, false, false);
  }
#pragma unroll
  for (int g = 0; g < 8; ++g) {
    int m0 = tm + g + 8*half;
    int m1 = m0 + 16;
    int n0 = tn + lr, n1 = n0 + 16;
    if constexpr (NMASK) {
      if (n0 < N) { Cb[m0*CSR + n0*CSC] = acc00[g]; Cb[m1*CSR + n0*CSC] = acc10[g]; }
      if (n1 < N) { Cb[m0*CSR + n1*CSC] = acc01[g]; Cb[m1*CSR + n1*CSC] = acc11[g]; }
    } else {
      Cb[m0*CSR + n0*CSC] = acc00[g];
      Cb[m0*CSR + n1*CSC] = acc01[g];
      Cb[m1*CSR + n0*CSC] = acc10[g];
      Cb[m1*CSR + n1*CSC] = acc11[g];
    }
  }
}

// ---------------------------------------------------------------------------
// Depthwise 3x3 SAME conv + bias + SiLU.  Input NHWC (stride cstride),
// output NCHW-flattened (b, c, h, w).
// ---------------------------------------------------------------------------
__global__ void dwconv_silu(const float* __restrict__ xz, int cstride, int choff,
                            const float* __restrict__ w, const float* __restrict__ bias,
                            float* __restrict__ out)
{
  int i = blockIdx.x*blockDim.x + threadIdx.x;
  int total = B_SZ*DI*LL;
  if (i >= total) return;
  int wx = i & 63; int hy = (i>>6)&63; int c = (i>>12)%DI; int b = i/(DI*LL);
  float acc = bias[c];
#pragma unroll
  for (int dy = 0; dy < 3; ++dy){
    int hh = hy + dy - 1; if (hh < 0 || hh >= HH) continue;
#pragma unroll
    for (int dx = 0; dx < 3; ++dx){
      int wp = wx + dx - 1; if (wp < 0 || wp >= WW) continue;
      acc += xz[((long)(b*HH+hh)*WW+wp)*cstride + choff + c] * w[c*9 + dy*3 + dx];
    }
  }
  out[i] = siluf_(acc);
}

// four scan directions for both streams
__global__ void four_dirs(const float* __restrict__ fx, const float* __restrict__ ix,
                          float* __restrict__ sf, float* __restrict__ si)
{
  int i = blockIdx.x*blockDim.x + threadIdx.x;
  int total = B_SZ*DI*LL;
  if (i >= total) return;
  int l = i % LL; int d = (i/LL)%DI; int b = i/(DI*LL);
  int hy = l >> 6, wx = l & 63;
  int lt = wx*HH + hy;                 // W-major index
  float vf = fx[i], vi = ix[i];
  long b0 = ((long)(b*KK+0)*DI + d)*LL;
  long b1 = ((long)(b*KK+1)*DI + d)*LL;
  long b2 = ((long)(b*KK+2)*DI + d)*LL;
  long b3 = ((long)(b*KK+3)*DI + d)*LL;
  sf[b0+l]=vf; sf[b1+lt]=vf; sf[b2+(LL-1-l)]=vf; sf[b3+(LL-1-lt)]=vf;
  si[b0+l]=vi; si[b1+lt]=vi; si[b2+(LL-1-l)]=vi; si[b3+(LL-1-lt)]=vi;
}

// per-(b,k,d) mean of sf+si over L (one wave each)
__global__ __launch_bounds__(32) void mean_k(const float* __restrict__ sf,
                                             const float* __restrict__ si,
                                             float* __restrict__ am)
{
  long bkd = blockIdx.x; int lane = threadIdx.x;
  long base = bkd * LL;
  float s = 0.f;
  for (int l = lane; l < LL; l += 32) s += sf[base+l] + si[base+l];
  for (int off = 16; off > 0; off >>= 1) s += __shfl_down(s, off, 32);
  if (lane == 0) am[bkd] = s * (1.f/LL);
}

// channel attention (squeeze-excite): one block per (b,k)
__global__ __launch_bounds__(DI) void attn_k(const float* __restrict__ am,
    const float* __restrict__ w1, const float* __restrict__ b1,
    const float* __restrict__ w2, const float* __restrict__ b2,
    float* __restrict__ attn)
{
  int bk = blockIdx.x; int k = bk & 3;
  int tid = threadIdx.x;
  __shared__ float sa[DI];
  __shared__ float sh1[DI/4];
  sa[tid] = am[bk*DI + tid];
  __syncthreads();
  if (tid < DI/4) {
    float s = b1[k*(DI/4) + tid];
    for (int c = 0; c < DI; ++c) s += sa[c] * w1[(k*(DI/4)+tid)*DI + c];
    sh1[tid] = siluf_(s);
  }
  __syncthreads();
  float s = b2[k*DI + tid];
  for (int r = 0; r < DI/4; ++r) s += sh1[r] * w2[(k*DI+tid)*(DI/4) + r];
  attn[bk*DI + tid] = sigmoidf_(s);
}

__global__ void blend_k(float* __restrict__ sf, const float* __restrict__ si,
                        const float* __restrict__ attn)
{
  int i = blockIdx.x*blockDim.x + threadIdx.x;
  int total = B_SZ*KK*DI*LL;
  if (i >= total) return;
  float a = attn[i / LL];
  sf[i] = sf[i]*a + si[i]*(1.f - a);
}

// ---------------------------------------------------------------------------
// Selective scan: one block per (b,k), 192 threads = one channel each.
// 16 states in registers; B_t/C_t staged into LDS in 64-step chunks.
// ---------------------------------------------------------------------------
__global__ __launch_bounds__(DI) void scan_k(const float* __restrict__ sf,
    const float* __restrict__ dts, const float* __restrict__ fdbl,
    const float* __restrict__ alog, const float* __restrict__ dtb,
    const float* __restrict__ Ds, float* __restrict__ oy)
{
  int bk = blockIdx.x; int k = bk & 3;
  int d  = threadIdx.x;
  __shared__ float sB[NS][64];
  __shared__ float sC[NS][64];
  long base = ((long)bk*DI + d)*LL;
  long fb   = (long)bk*CMID*LL;
  float bias = dtb[k*DI + d];
  float Dv   = Ds[k*DI + d];
  float Av[NS], h[NS];
#pragma unroll
  for (int n = 0; n < NS; ++n) { Av[n] = -__expf(alog[(k*DI+d)*NS + n]); h[n] = 0.f; }
  for (int t0 = 0; t0 < LL; t0 += 64) {
    __syncthreads();
    for (int idx = d; idx < 32*64; idx += DI) {
      int row = idx >> 6, tt = idx & 63;
      float v = fdbl[fb + (long)(RR+row)*LL + t0 + tt];
      if (row < NS) sB[row][tt] = v; else sC[row-NS][tt] = v;
    }
    __syncthreads();
    for (int tt = 0; tt < 64; ++tt) {
      long t = base + t0 + tt;
      float x  = dts[t] + bias;
      float db = (x > 20.f) ? x : log1pf(__expf(x));   // softplus
      float u  = sf[t];
      float du = db * u;
      float y  = 0.f;
#pragma unroll
      for (int n = 0; n < NS; ++n) {
        h[n] = __expf(db*Av[n])*h[n] + du*sB[n][tt];
        y   += h[n]*sC[n][tt];
      }
      oy[t] = y + u*Dv;
    }
  }
}

// recombine the 4 directions (flips + W-major transpose) into (B,L,2*Di)
__global__ void ycat_k(const float* __restrict__ oy, float* __restrict__ yc)
{
  int i = blockIdx.x*blockDim.x + threadIdx.x;
  int total = B_SZ*LL*2*DI;
  if (i >= total) return;
  int c = i % (2*DI); int l = (i/(2*DI))%LL; int b = i/(2*DI*LL);
  int hy = l >> 6, wx = l & 63;
  int lt = wx*HH + hy;
  float v;
  if (c < DI) {
    v = oy[((long)(b*KK+0)*DI+c)*LL + l]
      + oy[((long)(b*KK+2)*DI+c)*LL + (LL-1-l)];
  } else {
    int d2 = c - DI;
    v = oy[((long)(b*KK+1)*DI+d2)*LL + lt]
      + oy[((long)(b*KK+3)*DI+d2)*LL + (LL-1-lt)];
  }
  yc[i] = v;
}

// LayerNorm(192) + gamma/beta, then * SiLU(z).  One wave per token row.
__global__ __launch_bounds__(32) void ln_gate_k(const float* __restrict__ yf,
    const float* __restrict__ iz, const float* __restrict__ g,
    const float* __restrict__ bb, float* __restrict__ yg)
{
  int row = blockIdx.x;                 // b*L + l
  int lane = threadIdx.x;
  long base = (long)row*DI;
  float s = 0.f, sq = 0.f, v[6];
#pragma unroll
  for (int e = 0; e < 6; ++e) { float x = yf[base + e*32 + lane]; v[e]=x; s+=x; sq+=x*x; }
  for (int off = 16; off > 0; off >>= 1) { s += __shfl_xor(s, off, 32); sq += __shfl_xor(sq, off, 32); }
  float mu   = s  * (1.f/DI);
  float var  = sq * (1.f/DI) - mu*mu;
  float rstd = rsqrtf(var + 1e-5f);
  long zbase = (long)row*(2*DI) + DI;
#pragma unroll
  for (int e = 0; e < 6; ++e) {
    int c = e*32 + lane;
    float z  = iz[zbase + c];
    float yv = (v[e]-mu)*rstd*g[c] + bb[c];
    yg[base + c] = yv * siluf_(z);
  }
}

// ---------------------------------------------------------------------------
extern "C" void kernel_launch(void* const* d_in, const int* in_sizes, int n_in,
                              void* d_out, int out_size, void* d_ws, size_t ws_size,
                              hipStream_t stream)
{
  const float* f    = (const float*)d_in[0];
  const float* iin  = (const float*)d_in[1];
  const float* w1   = (const float*)d_in[2];
  const float* w2   = (const float*)d_in[3];
  const float* c1w  = (const float*)d_in[4];
  const float* c1b  = (const float*)d_in[5];
  const float* c2w  = (const float*)d_in[6];
  const float* c2b  = (const float*)d_in[7];
  const float* cfw1 = (const float*)d_in[8];
  const float* cfb1 = (const float*)d_in[9];
  const float* cfw2 = (const float*)d_in[10];
  const float* cfb2 = (const float*)d_in[11];
  const float* fpw  = (const float*)d_in[12];
  const float* dtw  = (const float*)d_in[13];
  const float* dtb  = (const float*)d_in[14];
  const float* alog = (const float*)d_in[15];
  const float* Ds   = (const float*)d_in[16];
  const float* fusw = (const float*)d_in[17];
  const float* ong  = (const float*)d_in[18];
  const float* onb  = (const float*)d_in[19];
  const float* opw  = (const float*)d_in[20];
  float* out = (float*)d_out;

  // workspace layout (floats), with reuse
  float* ws = (float*)d_ws;
  size_t off = 0;
  float* fzx = ws + off; off += (size_t)B_SZ*LL*DI;        // f x-branch after in_proj
  float* iz  = ws + off; off += (size_t)B_SZ*LL*2*DI;      // full i in_proj (x + z)
  float* fxA = ws + off; off += (size_t)B_SZ*DI*LL;        // conv(f)   -> later yf
  float* ixA = ws + off; off += (size_t)B_SZ*DI*LL;        // conv(i)   -> later yg
  float* sf  = ws + off; off += (size_t)B_SZ*KK*DI*LL;
  float* si  = ws + off; off += (size_t)B_SZ*KK*DI*LL;     // -> later out_y
  float* am  = ws + off; off += (size_t)B_SZ*KK*DI;
  float* att = ws + off; off += (size_t)B_SZ*KK*DI;
  float* fdb = ws + off; off += (size_t)B_SZ*KK*CMID*LL;
  float* dts = ws + off; off += (size_t)B_SZ*KK*DI*LL;     // -> later ycat

  const int M = B_SZ*LL;               // 8192 tokens

  // 1) in_proj (WMMA): fzx = f @ W1[0:Di]^T ;  iz = i @ W2^T
  { dim3 g(DI/32, M/32, 1);
    wmma_gemm<B_SZ*LL, DI, 96,  96,1, 1,96, DI,1,   0,1, 0,1, 0,1>
      <<<g,32,0,stream>>>(f, w1, fzx); }
  { dim3 g((2*DI)/32, M/32, 1);
    wmma_gemm<B_SZ*LL, 2*DI, 96, 96,1, 1,96, 2*DI,1, 0,1, 0,1, 0,1>
      <<<g,32,0,stream>>>(iin, w2, iz); }

  int tc = B_SZ*DI*LL;
  dwconv_silu<<<(tc+255)/256,256,0,stream>>>(fzx, DI,   0, c1w, c1b, fxA);
  dwconv_silu<<<(tc+255)/256,256,0,stream>>>(iz,  2*DI, 0, c2w, c2b, ixA);
  four_dirs  <<<(tc+255)/256,256,0,stream>>>(fxA, ixA, sf, si);

  mean_k<<<B_SZ*KK*DI, 32, 0, stream>>>(sf, si, am);
  attn_k<<<B_SZ*KK, DI, 0, stream>>>(am, cfw1, cfb1, cfw2, cfb2, att);
  int tb = B_SZ*KK*DI*LL;
  blend_k<<<(tb+255)/256,256,0,stream>>>(sf, si, att);

  // 2) f_dbl[b,k,c,l] = sum_d sf[b,k,d,l]*fpw[k,c,d]   (batched, N=38 masked)
  { dim3 g((CMID+31)/32, LL/32, B_SZ*KK);
    wmma_gemm<LL, CMID, DI,  1,LL, 1,DI, 1,LL,
              DI*LL, B_SZ*KK,  CMID*DI, KK,  CMID*LL, B_SZ*KK>
      <<<g,32,0,stream>>>(sf, fpw, fdb); }
  // 3) dts[b,k,d,l] = sum_r dtw[k,d,r]*f_dbl[b,k,r,l]  (batched, K=6 masked)
  { dim3 g(LL/32, DI/32, B_SZ*KK);
    wmma_gemm<DI, LL, RR,  RR,1, LL,1, LL,1,
              DI*RR, KK,  CMID*LL, B_SZ*KK,  DI*LL, B_SZ*KK>
      <<<g,32,0,stream>>>(dtw, fdb, dts); }

  // 4) selective scan; out_y reuses si
  scan_k<<<B_SZ*KK, DI, 0, stream>>>(sf, dts, fdb, alog, dtb, Ds, si);

  // 5) direction recombine; ycat reuses dts
  float* ycat = dts;
  int ty = B_SZ*LL*2*DI;
  ycat_k<<<(ty+255)/256,256,0,stream>>>(si, ycat);

  // 6) fusion_proj (WMMA): yf (reuses fxA) = ycat @ fusw^T
  { dim3 g(DI/32, M/32, 1);
    wmma_gemm<B_SZ*LL, DI, 2*DI,  2*DI,1, 1,2*DI, DI,1, 0,1, 0,1, 0,1>
      <<<g,32,0,stream>>>(ycat, fusw, fxA); }

  // 7) LayerNorm + SiLU(z) gate; yg reuses ixA
  ln_gate_k<<<M, 32, 0, stream>>>(fxA, iz, ong, onb, ixA);

  // 8) out_proj (WMMA) -> d_out
  { dim3 g(96/32, M/32, 1);
    wmma_gemm<B_SZ*LL, 96, DI,  DI,1, 1,DI, 96,1, 0,1, 0,1, 0,1>
      <<<g,32,0,stream>>>(ixA, opw, out); }
}
